// FisherAxialBlock_86234353369602
// MI455X (gfx1250) — matrix operand order repaired and checked by reading, not compiled, not run on hardware
//
#include <hip/hip_runtime.h>
#include <hip/hip_bf16.h>
#include <stdint.h>

typedef __attribute__((ext_vector_type(16))) _Float16 v16h;
typedef __attribute__((ext_vector_type(8)))  float    v8f;
typedef __attribute__((ext_vector_type(4)))  uint32_t v4u;

union Frag {
  v16h h;
  uint32_t u[8];
  v4u q[2];
};

#define D 64
#define CELLS 81
#define ZELEM (CELLS * D)   // 5184
#define NTHREADS 192

// d_ws dword layout: packed WMMA B-fragments (f16 pairs) for all weights
#define W1OFF 0             // 3 mlps * 18 ksteps * 4 ntiles * 32 lanes * 8 dwords = 55296
#define W2OFF 55296         // 3 * 2 * 4 * 32 * 8 = 6144
#define GOFF  61440         // 6 * 4 * 32 * 8 = 6144
#define WTOTAL 67584        // dwords (270336 bytes)

__device__ __forceinline__ uint32_t packf16(float a, float b) {
  union { _Float16 h[2]; uint32_t u; } p;
  p.h[0] = (_Float16)a;
  p.h[1] = (_Float16)b;
  return p.u;
}

// context permutation: MLP m, context-row `row` (0..8), concat slot j (0..8) -> cell index
__device__ __forceinline__ int perm_cell(int m, int row, int j) {
  if (m == 0) return row * 9 + j;            // row context
  if (m == 1) return j * 9 + row;            // col context
  int br = row / 3, bc = row - br * 3;       // box context
  int r = j / 3,   c = j - r * 3;
  return (br * 3 + r) * 9 + (bc * 3 + c);
}

// ---------------- weight fragment packing (runs once per launch, tiny) ----------------
__global__ void pack_weights(const float* rw1, const float* cw1, const float* bw1,
                             const float* rw2, const float* cw2, const float* bw2,
                             const float* gw, uint32_t* wf) {
  int i = blockIdx.x * blockDim.x + threadIdx.x;
  if (i >= WTOTAL) return;
  const float* w;
  int ks, rem2;
  if (i < W2OFF) {                       // w1 fragments, K=576
    int m = i / 18432, rem = i % 18432;
    ks = rem / 1024; rem2 = rem % 1024;
    w = (m == 0) ? rw1 : ((m == 1) ? cw1 : bw1);
  } else if (i < GOFF) {                 // w2 fragments, K=64
    int j = i - W2OFF;
    int m = j / 2048, rem = j % 2048;
    ks = rem / 1024; rem2 = rem % 1024;
    w = (m == 0) ? rw2 : ((m == 1) ? cw2 : bw2);
  } else {                               // gate_w fragments, K=192
    int j = i - GOFF;
    ks = j / 1024; rem2 = j % 1024;
    w = gw;
  }
  int nt   = rem2 / 256;
  int rem3 = rem2 % 256;
  int lane = rem3 / 8;
  int v    = rem3 % 8;
  int n = nt * 16 + (lane & 15);
  int k = ks * 32 + (v >> 2) * 16 + (lane >> 4) * 8 + 2 * (v & 3);   // B layout: lanes0-15 K 0..7,16..23, lanes16-31 K 8..15,24..31
  wf[i] = packf16(w[k * D + n], w[(k + 1) * D + n]);
}

// ---------------- main fused kernel: one block per batch sample ----------------
__global__ __launch_bounds__(NTHREADS, 1)
void fisher_axial_kernel(const float* __restrict__ z,
                         const uint32_t* __restrict__ wf,
                         const float* __restrict__ rb1, const float* __restrict__ rb2,
                         const float* __restrict__ cb1, const float* __restrict__ cb2,
                         const float* __restrict__ bb1, const float* __restrict__ bb2,
                         const float* __restrict__ gb,  const float* __restrict__ alphap,
                         float* __restrict__ out) {
  __shared__ __align__(16) float    zf32[ZELEM];       // 20736 B
  __shared__ __align__(16) _Float16 zf16[ZELEM];       // 10368 B
  __shared__ __align__(16) _Float16 hbuf[3 * 16 * D];  //  6144 B (hidden, padded to 16 rows)
  __shared__ __align__(16) _Float16 ubuf[3 * 16 * D];  //  6144 B (U_row / U_col / U_box)
  __shared__ __align__(16) _Float16 boxfull[9 * 576];  // 10368 B (scrambled box broadcast)

  const int b    = blockIdx.x;
  const int tid  = threadIdx.x;
  const int wave = tid >> 5;
  const int lane = tid & 31;
  const int mrow = lane & 15;
  const int hi   = lane >> 4;
  const float* zb = z + (size_t)b * ZELEM;

  // ---- stage z into LDS (f32 for residual, f16 for WMMA A operands) ----
  for (int i = tid; i < ZELEM; i += NTHREADS) {
    float v = zb[i];
    zf32[i] = v;
    zf16[i] = (_Float16)v;
  }
  __syncthreads();

  // ---- phase 1: layer-1 of the three context MLPs (wave m owns MLP m) ----
  if (wave < 3) {
    const int m = wave;
    const float* b1 = (m == 0) ? rb1 : ((m == 1) ? cb1 : bb1);
    v8f acc0{}, acc1{}, acc2{}, acc3{};
    const int r9 = (mrow < 9) ? mrow : 8;  // rows 9..15 are padding; clamp loads
    for (int ks = 0; ks < 18; ++ks) {
      const int j = ks >> 1, hf = ks & 1;
      const int cell = perm_cell(m, r9, j);
      const uint32_t* ap = (const uint32_t*)(zf16 + cell * D + hf * 32);
      Frag a;
#pragma unroll
      for (int v = 0; v < 4; ++v) a.u[v] = ap[hi * 4 + v];
#pragma unroll
      for (int v = 0; v < 4; ++v) a.u[4 + v] = ap[8 + hi * 4 + v];
      const uint32_t* wp = wf + W1OFF + ((m * 18 + ks) * 4) * 256 + lane * 8;
      Frag b0, b1f, b2f, b3f;
      b0.q[0]  = ((const v4u*)(wp        ))[0]; b0.q[1]  = ((const v4u*)(wp        ))[1];
      b1f.q[0] = ((const v4u*)(wp + 256  ))[0]; b1f.q[1] = ((const v4u*)(wp + 256  ))[1];
      b2f.q[0] = ((const v4u*)(wp + 512  ))[0]; b2f.q[1] = ((const v4u*)(wp + 512  ))[1];
      b3f.q[0] = ((const v4u*)(wp + 768  ))[0]; b3f.q[1] = ((const v4u*)(wp + 768  ))[1];
      acc0 = __builtin_amdgcn_wmma_f32_16x16x32_f16(false, a.h, false, b0.h,  (short)0, acc0, false, false);
      acc1 = __builtin_amdgcn_wmma_f32_16x16x32_f16(false, a.h, false, b1f.h, (short)0, acc1, false, false);
      acc2 = __builtin_amdgcn_wmma_f32_16x16x32_f16(false, a.h, false, b2f.h, (short)0, acc2, false, false);
      acc3 = __builtin_amdgcn_wmma_f32_16x16x32_f16(false, a.h, false, b3f.h, (short)0, acc3, false, false);
    }
    // bias + exact GELU, write hidden to LDS (C layout: N = nt*16 + (lane&15), M = r + 8*hi)
    v8f accs[4] = {acc0, acc1, acc2, acc3};
#pragma unroll
    for (int nt = 0; nt < 4; ++nt) {
      const int n = nt * 16 + (lane & 15);
      const float bias = b1[n];
#pragma unroll
      for (int r = 0; r < 8; ++r) {
        const int Mr = r + 8 * hi;
        float x = accs[nt][r] + bias;
        float g = 0.5f * x * (1.0f + erff(x * 0.70710678118654752f));
        hbuf[m * (16 * D) + Mr * D + n] = (_Float16)g;
      }
    }
  }
  __syncthreads();

  // ---- phase 2: layer-2 (16x64 @ 64x64) ----
  if (wave < 3) {
    const int m = wave;
    const float* b2 = (m == 0) ? rb2 : ((m == 1) ? cb2 : bb2);
    v8f acc0{}, acc1{}, acc2{}, acc3{};
    for (int ks = 0; ks < 2; ++ks) {
      const uint32_t* ap = (const uint32_t*)(hbuf + m * (16 * D) + mrow * D + ks * 32);
      Frag a;
#pragma unroll
      for (int v = 0; v < 4; ++v) a.u[v] = ap[hi * 4 + v];
#pragma unroll
      for (int v = 0; v < 4; ++v) a.u[4 + v] = ap[8 + hi * 4 + v];
      const uint32_t* wp = wf + W2OFF + ((m * 2 + ks) * 4) * 256 + lane * 8;
      Frag b0, b1f, b2f, b3f;
      b0.q[0]  = ((const v4u*)(wp        ))[0]; b0.q[1]  = ((const v4u*)(wp        ))[1];
      b1f.q[0] = ((const v4u*)(wp + 256  ))[0]; b1f.q[1] = ((const v4u*)(wp + 256  ))[1];
      b2f.q[0] = ((const v4u*)(wp + 512  ))[0]; b2f.q[1] = ((const v4u*)(wp + 512  ))[1];
      b3f.q[0] = ((const v4u*)(wp + 768  ))[0]; b3f.q[1] = ((const v4u*)(wp + 768  ))[1];
      acc0 = __builtin_amdgcn_wmma_f32_16x16x32_f16(false, a.h, false, b0.h,  (short)0, acc0, false, false);
      acc1 = __builtin_amdgcn_wmma_f32_16x16x32_f16(false, a.h, false, b1f.h, (short)0, acc1, false, false);
      acc2 = __builtin_amdgcn_wmma_f32_16x16x32_f16(false, a.h, false, b2f.h, (short)0, acc2, false, false);
      acc3 = __builtin_amdgcn_wmma_f32_16x16x32_f16(false, a.h, false, b3f.h, (short)0, acc3, false, false);
    }
    v8f accs[4] = {acc0, acc1, acc2, acc3};
#pragma unroll
    for (int nt = 0; nt < 4; ++nt) {
      const int n = nt * 16 + (lane & 15);
      const float bias = b2[n];
#pragma unroll
      for (int r = 0; r < 8; ++r) {
        const int Mr = r + 8 * hi;
        ubuf[m * (16 * D) + Mr * D + n] = (_Float16)(accs[nt][r] + bias);
      }
    }
  }
  __syncthreads();

  // ---- phase 3: materialize the torch expand+reshape scramble of box_up ----
  // boxfull[a1][x] = U_box[a1][ ((x) % 192) / 3 ],  x = a2*64 + dn
  for (int i = tid; i < 9 * 576; i += NTHREADS) {
    const int a1 = i / 576;
    const int x  = i - a1 * 576;
    const int d  = (x % 192) / 3;
    boxfull[i] = ubuf[2 * (16 * D) + a1 * D + d];
  }
  __syncthreads();

  // ---- phase 4: gate matmul (81x192 @ 192x64), each wave owns one 16-row M-tile ----
  {
    const int tile = wave;                       // 6 waves x 16 rows covers 81..95
    int cellc = tile * 16 + mrow;
    if (cellc > 80) cellc = 80;                  // clamp loads for padding rows
    const int a1c = cellc / 9, a2c = cellc - a1c * 9;
    v8f acc0{}, acc1{}, acc2{}, acc3{};
    for (int ks = 0; ks < 6; ++ks) {
      const int seg = ks >> 1, hf = ks & 1;
      const _Float16* base;
      if (seg == 0)      base = ubuf + a1c * D + hf * 32;                        // row_upb
      else if (seg == 1) base = ubuf + (16 * D) + a2c * D + hf * 32;             // col_upb
      else               base = boxfull + a1c * 576 + a2c * D + hf * 32;         // box (scrambled)
      const uint32_t* ap = (const uint32_t*)base;
      Frag a;
#pragma unroll
      for (int v = 0; v < 4; ++v) a.u[v] = ap[hi * 4 + v];
#pragma unroll
      for (int v = 0; v < 4; ++v) a.u[4 + v] = ap[8 + hi * 4 + v];
      const uint32_t* wp = wf + GOFF + (ks * 4) * 256 + lane * 8;
      Frag b0, b1f, b2f, b3f;
      b0.q[0]  = ((const v4u*)(wp        ))[0]; b0.q[1]  = ((const v4u*)(wp        ))[1];
      b1f.q[0] = ((const v4u*)(wp + 256  ))[0]; b1f.q[1] = ((const v4u*)(wp + 256  ))[1];
      b2f.q[0] = ((const v4u*)(wp + 512  ))[0]; b2f.q[1] = ((const v4u*)(wp + 512  ))[1];
      b3f.q[0] = ((const v4u*)(wp + 768  ))[0]; b3f.q[1] = ((const v4u*)(wp + 768  ))[1];
      acc0 = __builtin_amdgcn_wmma_f32_16x16x32_f16(false, a.h, false, b0.h,  (short)0, acc0, false, false);
      acc1 = __builtin_amdgcn_wmma_f32_16x16x32_f16(false, a.h, false, b1f.h, (short)0, acc1, false, false);
      acc2 = __builtin_amdgcn_wmma_f32_16x16x32_f16(false, a.h, false, b2f.h, (short)0, acc2, false, false);
      acc3 = __builtin_amdgcn_wmma_f32_16x16x32_f16(false, a.h, false, b3f.h, (short)0, acc3, false, false);
    }
    // sigmoid gate + gated residual epilogue
    const float alpha = alphap[0];
    v8f accs[4] = {acc0, acc1, acc2, acc3};
#pragma unroll
    for (int nt = 0; nt < 4; ++nt) {
      const int n0 = nt * 16 + (lane & 15);
      const float gbias = gb[n0];
#pragma unroll
      for (int r = 0; r < 8; ++r) {
        const int Mr = r + 8 * hi;
        const int cell = tile * 16 + Mr;
        if (cell < CELLS) {
          const int a1 = cell / 9, a2 = cell - a1 * 9;
          float x = accs[nt][r] + gbias;
          float g = 1.0f / (1.0f + expf(-x));
          float dlt = (float)ubuf[a1 * D + n0]
                    + (float)ubuf[(16 * D) + a2 * D + n0]
                    + (float)boxfull[a1 * 576 + a2 * D + n0];
          const int idx = cell * D + n0;
          out[(size_t)b * ZELEM + idx] = zf32[idx] + alpha * g * dlt;
        }
      }
    }
  }
}

extern "C" void kernel_launch(void* const* d_in, const int* in_sizes, int n_in,
                              void* d_out, int out_size, void* d_ws, size_t ws_size,
                              hipStream_t stream) {
  const float* z   = (const float*)d_in[0];
  const float* rw1 = (const float*)d_in[1];  const float* rb1 = (const float*)d_in[2];
  const float* rw2 = (const float*)d_in[3];  const float* rb2 = (const float*)d_in[4];
  const float* cw1 = (const float*)d_in[5];  const float* cb1 = (const float*)d_in[6];
  const float* cw2 = (const float*)d_in[7];  const float* cb2 = (const float*)d_in[8];
  const float* bw1 = (const float*)d_in[9];  const float* bb1 = (const float*)d_in[10];
  const float* bw2 = (const float*)d_in[11]; const float* bb2 = (const float*)d_in[12];
  const float* gw  = (const float*)d_in[13]; const float* gb  = (const float*)d_in[14];
  const float* alpha = (const float*)d_in[15];
  uint32_t* wf = (uint32_t*)d_ws;
  float* out = (float*)d_out;

  const int Bv = in_sizes[0] / ZELEM;

  pack_weights<<<(WTOTAL + 255) / 256, 256, 0, stream>>>(rw1, cw1, bw1, rw2, cw2, bw2, gw, wf);
  fisher_axial_kernel<<<Bv, NTHREADS, 0, stream>>>(z, wf, rb1, rb2, cb1, cb2, bb1, bb2,
                                                   gb, alpha, out);
}
